// DeepseekV32Indexer_87832081203520
// MI455X (gfx1250) — compile-verified, matrix operand order checked by reading, not run on hardware
//
#include <hip/hip_runtime.h>
#include <hip/hip_bf16.h>
#include <cstdint>

// ---------------------------------------------------------------------------
// DeepSeek V3.2 indexer for MI455X (gfx1250), wave32 + WMMA bf16 + TDM.
// B=1, S=2048, HID=7168, QLR=1536, NH=64, HD=128, RD=64, TOPK=1024
// ---------------------------------------------------------------------------

#define S_   2048
#define HID_ 7168
#define QLR_ 1536
#define NH_  64
#define HD_  128
#define NHD_ (NH_ * HD_)   // 8192
#define TOPK_ 1024

typedef __bf16 bf16_t;
typedef __attribute__((ext_vector_type(8)))  __bf16 v8bf;
typedef __attribute__((ext_vector_type(16))) __bf16 v16bf;
typedef __attribute__((ext_vector_type(8)))  float  v8f;

typedef uint32_t u32x4 __attribute__((ext_vector_type(4)));
typedef int32_t  i32x4 __attribute__((ext_vector_type(4)));
typedef int32_t  i32x8 __attribute__((ext_vector_type(8)));

#if defined(__has_builtin)
#if __has_builtin(__builtin_amdgcn_tensor_load_to_lds) && \
    __has_builtin(__builtin_amdgcn_s_wait_tensorcnt)
#define HAVE_TDM 1
#endif
#endif
#ifndef HAVE_TDM
#define HAVE_TDM 0
#endif

#define FWHT_SCALE 0.08838834764831845f   // 128^-0.5 (also used as HD^-0.5)

// Load a 16x32 (or 32x16) WMMA bf16 fragment chunk for one lane.
// `base` points at this lane's row/col start of the 32-wide K range.
// Layout (ISA 7.12.2): elem e<8 -> K = 8*hi + e ; e>=8 -> K = 16 + 8*hi + (e-8)
template <typename P>
__device__ inline v16bf load_frag(const P* base, int hi) {
  v8bf c0 = *reinterpret_cast<const v8bf*>(base + 8 * hi);
  v8bf c1 = *reinterpret_cast<const v8bf*>(base + 16 + 8 * hi);
  v16bf f;
#pragma unroll
  for (int e = 0; e < 8; ++e) { f[e] = c0[e]; f[e + 8] = c1[e]; }
  return f;
}

// --------------------------- conversion kernels ----------------------------

__global__ __launch_bounds__(256) void cvt_bf16_kernel(const float* __restrict__ in,
                                                       bf16_t* __restrict__ out, int n) {
  int i = blockIdx.x * 256 + threadIdx.x;
  if (i < n) out[i] = (bf16_t)in[i];
}

// wq_b [QLR][NHD] f32  ->  wqT [NHD][QLR] bf16
__global__ __launch_bounds__(256) void cvt_wqT_kernel(const float* __restrict__ in,
                                                      bf16_t* __restrict__ out) {
  int i = blockIdx.x * 256 + threadIdx.x;
  if (i < QLR_ * NHD_) {
    int k = i / NHD_;
    int n = i - k * NHD_;
    out[(size_t)n * QLR_ + k] = (bf16_t)in[i];
  }
}

// ------------------- k path: x@wk + LN + RoPE + FWHT, fused w = x@ww -------

__global__ __launch_bounds__(128) void kproj_kernel(
    const float* __restrict__ x, const float* __restrict__ wk,
    const float* __restrict__ wwt, const float* __restrict__ gamma,
    const float* __restrict__ beta, const float* __restrict__ cosb,
    const float* __restrict__ sinb, bf16_t* __restrict__ Kb,
    float* __restrict__ Wout) {
  __shared__ float xs[8][512];
  __shared__ float red[128];
  __shared__ float row[128];
  const int d = threadIdx.x;              // 0..127 = k dim
  const int tok0 = blockIdx.x * 8;

  float kacc[8], wacc[8];
#pragma unroll
  for (int r = 0; r < 8; ++r) { kacc[r] = 0.f; wacc[r] = 0.f; }

  for (int i0 = 0; i0 < HID_; i0 += 512) {
    for (int i = d; i < 8 * 512; i += 128) {
      int r = i >> 9, ii = i & 511;
      xs[r][ii] = x[(size_t)(tok0 + r) * HID_ + i0 + ii];
    }
    __syncthreads();
    for (int ii = 0; ii < 512; ++ii) {
      float kw = wk[(size_t)(i0 + ii) * HD_ + d];
      float ww = wwt[(size_t)(i0 + ii) * NH_ + (d & 63)];
#pragma unroll
      for (int r = 0; r < 8; ++r) {
        float xv = xs[r][ii];
        kacc[r] += xv * kw;
        wacc[r] += xv * ww;
      }
    }
    __syncthreads();
  }

  if (d < NH_) {
#pragma unroll
    for (int r = 0; r < 8; ++r)
      Wout[(size_t)(tok0 + r) * NH_ + d] = wacc[r] * 0.125f;  // NH^-0.5
  }

  const float g = gamma[d], bt = beta[d];
  for (int r = 0; r < 8; ++r) {
    const int tok = tok0 + r;
    // ---- layernorm over 128 dims ----
    red[d] = kacc[r]; __syncthreads();
    for (int off = 64; off > 0; off >>= 1) {
      if (d < off) red[d] += red[d + off];
      __syncthreads();
    }
    float mu = red[0] * (1.f / 128.f); __syncthreads();
    float dv = kacc[r] - mu;
    red[d] = dv * dv; __syncthreads();
    for (int off = 64; off > 0; off >>= 1) {
      if (d < off) red[d] += red[d + off];
      __syncthreads();
    }
    float var = red[0] * (1.f / 128.f); __syncthreads();
    float kn = dv * rsqrtf(var + 1e-6f) * g + bt;
    // ---- rope (pairs (d, d+32) for d<64) ----
    row[d] = kn; __syncthreads();
    float o;
    if (d < 32) {
      float c = cosb[tok * 32 + d], s = sinb[tok * 32 + d];
      o = row[d] * c - row[d + 32] * s;
    } else if (d < 64) {
      int dd = d - 32;
      float c = cosb[tok * 32 + dd], s = sinb[tok * 32 + dd];
      o = row[dd] * s + row[d] * c;
    } else {
      o = kn;
    }
    __syncthreads(); row[d] = o; __syncthreads();
    // ---- FWHT(128) ----
    for (int hs = 1; hs < 128; hs <<= 1) {
      float a = row[d], b = row[d ^ hs];
      __syncthreads();
      row[d] = (d & hs) ? (b - a) : (a + b);
      __syncthreads();
    }
    Kb[(size_t)tok * HD_ + d] = (bf16_t)(row[d] * FWHT_SCALE);
  }
}

// ------------------- q GEMM: qr_b[2048][1536] @ wqT[8192][1536]^T ----------
// Two-deep software pipeline with compile-time buffer selection (no runtime
// array indexing -> fragments stay in VGPRs, no scratch spills).

__global__ __launch_bounds__(256) void qgemm_kernel(const bf16_t* __restrict__ A,
                                                    const bf16_t* __restrict__ BT,
                                                    bf16_t* __restrict__ Qout) {
  const int tid = threadIdx.x;
  const int wave = tid >> 5, lane = tid & 31;
  const int m = lane & 15, hi = lane >> 4;
  const int t0 = blockIdx.y * 16;
  const int n0 = blockIdx.x * 512 + wave * 64;   // wave tile: 16 x 64

  const bf16_t* arow = A + (size_t)(t0 + m) * QLR_;
  const bf16_t* brow0 = BT + (size_t)(n0 + 0 * 16 + m) * QLR_;
  const bf16_t* brow1 = BT + (size_t)(n0 + 1 * 16 + m) * QLR_;
  const bf16_t* brow2 = BT + (size_t)(n0 + 2 * 16 + m) * QLR_;
  const bf16_t* brow3 = BT + (size_t)(n0 + 3 * 16 + m) * QLR_;

  v8f acc[4];
#pragma unroll
  for (int nt = 0; nt < 4; ++nt) acc[nt] = (v8f){};

  v16bf a0, a1, b0[4], b1[4];
  a0 = load_frag(arow, hi);
  b0[0] = load_frag(brow0, hi);
  b0[1] = load_frag(brow1, hi);
  b0[2] = load_frag(brow2, hi);
  b0[3] = load_frag(brow3, hi);

  for (int k0 = 0; k0 < QLR_; k0 += 64) {   // QLR_ % 64 == 0
    // prefetch next chunk (k0+32) into buffer 1 while buffer 0 computes
    __builtin_prefetch(arow + k0 + 96, 0, 3);
    a1 = load_frag(arow + k0 + 32, hi);
    b1[0] = load_frag(brow0 + k0 + 32, hi);
    b1[1] = load_frag(brow1 + k0 + 32, hi);
    b1[2] = load_frag(brow2 + k0 + 32, hi);
    b1[3] = load_frag(brow3 + k0 + 32, hi);
#pragma unroll
    for (int nt = 0; nt < 4; ++nt)
      acc[nt] = __builtin_amdgcn_wmma_f32_16x16x32_bf16(
          false, a0, false, b0[nt], (short)0, acc[nt], false, false);
    if (k0 + 64 < QLR_) {
      a0 = load_frag(arow + k0 + 64, hi);
      b0[0] = load_frag(brow0 + k0 + 64, hi);
      b0[1] = load_frag(brow1 + k0 + 64, hi);
      b0[2] = load_frag(brow2 + k0 + 64, hi);
      b0[3] = load_frag(brow3 + k0 + 64, hi);
    }
#pragma unroll
    for (int nt = 0; nt < 4; ++nt)
      acc[nt] = __builtin_amdgcn_wmma_f32_16x16x32_bf16(
          false, a1, false, b1[nt], (short)0, acc[nt], false, false);
  }
#pragma unroll
  for (int nt = 0; nt < 4; ++nt)
#pragma unroll
    for (int r = 0; r < 8; ++r)
      Qout[(size_t)(t0 + r + 8 * hi) * NHD_ + n0 + nt * 16 + m] = (bf16_t)acc[nt][r];
}

// ---------------- q epilogue: per-(t,h) RoPE + FWHT, in place --------------

__global__ __launch_bounds__(128) void qpost_kernel(bf16_t* __restrict__ Q,
                                                    const float* __restrict__ cosb,
                                                    const float* __restrict__ sinb) {
  __shared__ float row[128];
  const int bx = blockIdx.x;           // t*64 + h
  const int t = bx >> 6;
  const int d = threadIdx.x;
  bf16_t* p = Q + (size_t)bx * HD_;
  float v = (float)p[d];
  row[d] = v; __syncthreads();
  float o;
  if (d < 32) {
    float c = cosb[t * 32 + d], s = sinb[t * 32 + d];
    o = row[d] * c - row[d + 32] * s;
  } else if (d < 64) {
    int dd = d - 32;
    float c = cosb[t * 32 + dd], s = sinb[t * 32 + dd];
    o = row[dd] * s + row[d] * c;
  } else {
    o = v;
  }
  __syncthreads(); row[d] = o; __syncthreads();
  for (int hs = 1; hs < 128; hs <<= 1) {
    float a = row[d], b = row[d ^ hs];
    __syncthreads();
    row[d] = (d & hs) ? (b - a) : (a + b);
    __syncthreads();
  }
  p[d] = (bf16_t)(row[d] * FWHT_SCALE);
}

// ---- score: per head relu(Q_h K^T), head-weighted sum, scale, causal mask --
// Block = 128 threads (4 waves), tile 16t x 128s. The 32KB K-slice for the
// block is staged into LDS with one TDM tensor_load_to_lds; B fragments are
// then read once from LDS and held in registers across all 64 heads. A
// fragments (Q) use a two-deep pipeline with static buffer selection.

__global__ __launch_bounds__(128) void score_kernel(const bf16_t* __restrict__ Q,
                                                    const bf16_t* __restrict__ Kb,
                                                    const float* __restrict__ W,
                                                    float* __restrict__ out) {
  __shared__ float wsh[16 * NH_];          // 4 KB: head weights for this t-tile
  __shared__ bf16_t ksh[128 * HD_];        // 32 KB: K rows [sblk, sblk+128)
  const int tid = threadIdx.x;
  const int wave = tid >> 5, lane = tid & 31;
  const int n = lane & 15, hi = lane >> 4;
  const int t0 = blockIdx.y * 16;
  const int sblk = blockIdx.x * 128;
  const int s0 = sblk + wave * 32;         // wave tile: 16t x 32s

  for (int i = tid; i < 16 * NH_; i += 128) {
    int r = i >> 6, h = i & 63;
    wsh[i] = W[(size_t)(t0 + r) * NH_ + h];
  }

#if HAVE_TDM
  if (tid < 32) {
    // D# for a 2D tile: data_size=8B, row = 32 units, 128 rows.
    uint64_t ga = (uint64_t)(uintptr_t)(Kb + (size_t)sblk * HD_);
    uint32_t lds_off = (uint32_t)(uintptr_t)ksh;
    u32x4 g0 = { 1u,                                   // count=1, user mode
                 lds_off,                              // lds_addr
                 (uint32_t)ga,                         // global_addr[31:0]
                 (uint32_t)((ga >> 32) & 0x1FFFFFFu) | (2u << 30) };  // [56:32] | type=2
    i32x8 g1 = { (int)(3u << 16),     // workgroup_mask=0, data_size=3 (8B)
                 (int)(32u << 16),    // tensor_dim0 = 32 (bits 79:48, low half)
                 (int)(2048u << 16),  // tensor_dim0 hi=0, tensor_dim1 = 2048
                 (int)(32u << 16),    // tensor_dim1 hi=0, tile_dim0 = 32
                 (int)128,            // tile_dim1 = 128, tile_dim2 = 0
                 (int)32,             // tensor_dim0_stride = 32
                 (int)(32u << 16),    // d0_stride hi=0, tensor_dim1_stride = 32
                 0 };
    i32x4 gz = { 0, 0, 0, 0 };
#if __clang_major__ >= 23
    i32x8 gz8 = { 0, 0, 0, 0, 0, 0, 0, 0 };
    __builtin_amdgcn_tensor_load_to_lds(g0, g1, gz, gz, gz8, 0);
#else
    __builtin_amdgcn_tensor_load_to_lds(g0, g1, gz, gz, 0);
#endif
    __builtin_amdgcn_s_wait_tensorcnt(0);
  }
#else
  for (int i = tid * 8; i < 128 * HD_; i += 128 * 8)
    *reinterpret_cast<v8bf*>(ksh + i) =
        *reinterpret_cast<const v8bf*>(Kb + (size_t)sblk * HD_ + i);
#endif
  __syncthreads();

  // B fragments from LDS: 2 s-subtiles x 4 K-chunks, reused across all heads
  v16bf bfrag[2][4];
#pragma unroll
  for (int st = 0; st < 2; ++st)
#pragma unroll
    for (int dk = 0; dk < 4; ++dk)
      bfrag[st][dk] =
          load_frag(ksh + (size_t)(wave * 32 + st * 16 + n) * HD_ + dk * 32, hi);

  float facc[2][8];
#pragma unroll
  for (int st = 0; st < 2; ++st)
#pragma unroll
    for (int r = 0; r < 8; ++r) facc[st][r] = 0.f;

  const bf16_t* qrow = Q + (size_t)(t0 + n) * NH_ * HD_;  // head-major rows

  // relu + head-weight accumulate for one head, fragments in `a`
  auto compute_head = [&](const v16bf (&a)[4], int h) {
#pragma unroll
    for (int st = 0; st < 2; ++st) {
      v8f c = (v8f){};
#pragma unroll
      for (int dk = 0; dk < 4; ++dk)
        c = __builtin_amdgcn_wmma_f32_16x16x32_bf16(
            false, a[dk], false, bfrag[st][dk], (short)0, c, false, false);
#pragma unroll
      for (int r = 0; r < 8; ++r) {
        float wv = wsh[(r + 8 * hi) * NH_ + h];
        float v = c[r];
        facc[st][r] += wv * (v > 0.f ? v : 0.f);
      }
    }
  };

  v16bf qa0[4], qa1[4];
#pragma unroll
  for (int dk = 0; dk < 4; ++dk) qa0[dk] = load_frag(qrow + dk * 32, hi);

  for (int h = 0; h < NH_; h += 2) {       // NH_ even
    __builtin_prefetch(qrow + (size_t)(h + 2) * HD_, 0, 3);
#pragma unroll
    for (int dk = 0; dk < 4; ++dk)
      qa1[dk] = load_frag(qrow + (size_t)(h + 1) * HD_ + dk * 32, hi);
    compute_head(qa0, h);
    if (h + 2 < NH_) {
#pragma unroll
      for (int dk = 0; dk < 4; ++dk)
        qa0[dk] = load_frag(qrow + (size_t)(h + 2) * HD_ + dk * 32, hi);
    }
    compute_head(qa1, h + 1);
  }

  const float NEG_INF = -__builtin_inff();
#pragma unroll
  for (int st = 0; st < 2; ++st)
#pragma unroll
    for (int r = 0; r < 8; ++r) {
      int t = t0 + r + 8 * hi;
      int s = s0 + st * 16 + n;
      float v = facc[st][r] * FWHT_SCALE;      // HD^-0.5
      if (s > t) v = NEG_INF;
      out[(size_t)t * S_ + s] = v;
    }
}

// ------------------ top-k via per-row bitonic sort in LDS ------------------

__global__ __launch_bounds__(256) void topk_kernel(const float* __restrict__ score,
                                                   int* __restrict__ idx_out) {
  __shared__ float vals[S_];
  __shared__ int   inds[S_];
  const int t = blockIdx.x, tid = threadIdx.x;
  for (int i = tid; i < S_; i += 256) {
    vals[i] = score[(size_t)t * S_ + i];
    inds[i] = i;
  }
  __syncthreads();
  for (int k2 = 2; k2 <= S_; k2 <<= 1) {
    for (int j = k2 >> 1; j > 0; j >>= 1) {
      for (int i = tid; i < S_; i += 256) {
        int ixj = i ^ j;
        if (ixj > i) {
          bool desc = (i & k2) == 0;
          float vi = vals[i], vj = vals[ixj];
          bool sw = desc ? (vi < vj) : (vi > vj);
          if (sw) {
            vals[i] = vj; vals[ixj] = vi;
            int ti = inds[i]; inds[i] = inds[ixj]; inds[ixj] = ti;
          }
        }
      }
      __syncthreads();
    }
  }
  for (int i = tid; i < TOPK_; i += 256) idx_out[(size_t)t * TOPK_ + i] = inds[i];
}

// ---------------------------------------------------------------------------

static inline size_t align256(size_t x) { return (x + 255) & ~(size_t)255; }

extern "C" void kernel_launch(void* const* d_in, const int* in_sizes, int n_in,
                              void* d_out, int out_size, void* d_ws, size_t ws_size,
                              hipStream_t stream) {
  const float* x    = (const float*)d_in[0];
  const float* qr   = (const float*)d_in[1];
  const float* cosb = (const float*)d_in[2];
  const float* sinb = (const float*)d_in[3];
  const float* wq_b = (const float*)d_in[4];
  const float* wk   = (const float*)d_in[5];
  const float* kg   = (const float*)d_in[6];
  const float* kb   = (const float*)d_in[7];
  const float* ww   = (const float*)d_in[8];
  // d_in[9] = index_topk scalar (=1024), baked in as TOPK_.

  float* score_out = (float*)d_out;
  int*   idx_out   = (int*)(score_out + (size_t)S_ * S_);

  char* ws = (char*)d_ws;
  size_t off = 0;
  bf16_t* Kb   = (bf16_t*)(ws + off); off = align256(off + (size_t)S_ * HD_ * 2);
  float*  W    = (float*) (ws + off); off = align256(off + (size_t)S_ * NH_ * 4);
  bf16_t* qr_b = (bf16_t*)(ws + off); off = align256(off + (size_t)S_ * QLR_ * 2);
  bf16_t* wqT  = (bf16_t*)(ws + off); off = align256(off + (size_t)NHD_ * QLR_ * 2);
  bf16_t* Q    = (bf16_t*)(ws + off); off = align256(off + (size_t)S_ * NHD_ * 2);
  (void)ws_size; (void)in_sizes; (void)n_in; (void)out_size;

  // 1. qr -> bf16
  cvt_bf16_kernel<<<(S_ * QLR_ + 255) / 256, 256, 0, stream>>>(qr, qr_b, S_ * QLR_);
  // 2. wq_b -> transposed bf16
  cvt_wqT_kernel<<<(QLR_ * NHD_ + 255) / 256, 256, 0, stream>>>(wq_b, wqT);
  // 3. k path (+ fused head weights)
  kproj_kernel<<<S_ / 8, 128, 0, stream>>>(x, wk, ww, kg, kb, cosb, sinb, Kb, W);
  // 4. q GEMM (WMMA bf16, two-deep pipeline)
  qgemm_kernel<<<dim3(NHD_ / 512, S_ / 16), 256, 0, stream>>>(qr_b, wqT, Q);
  // 5. q RoPE + FWHT
  qpost_kernel<<<S_ * NH_, 128, 0, stream>>>(Q, cosb, sinb);
  // 6. per-head relu score + head-weighted reduce + mask (WMMA bf16 + TDM)
  score_kernel<<<dim3(S_ / 128, S_ / 16), 128, 0, stream>>>(Q, Kb, W, score_out);
  // 7. top-k indices
  topk_kernel<<<S_, 256, 0, stream>>>(score_out, idx_out);
}
